// MemTransformerLM_43215960932414
// MI455X (gfx1250) — matrix operand order repaired and checked by reading, not compiled, not run on hardware
//
#include <hip/hip_runtime.h>

#define QLEN   2048
#define BSZn   4
#define NHEAD  16
#define DHEAD  64
#define DMODEL 1024
#define DINNER 4096

typedef __attribute__((ext_vector_type(16))) __bf16 v16bf;
typedef __attribute__((ext_vector_type(8)))  __bf16 v8bf;
typedef __attribute__((ext_vector_type(8)))  float  v8f;
typedef unsigned short u16b;

__device__ __forceinline__ v8f wmma_bf16(v16bf a, v16bf b, v8f c) {
  return __builtin_amdgcn_wmma_f32_16x16x32_bf16(false, a, false, b, (short)0, c,
                                                 false, false);
}

__device__ __forceinline__ v8f v8f_zero() {
  v8f z = {0.f, 0.f, 0.f, 0.f, 0.f, 0.f, 0.f, 0.f};
  return z;
}

__device__ __forceinline__ u16b f2bf(float x) {
  union { float f; unsigned int u; } cv; cv.f = x;
  unsigned int r = cv.u + 0x7FFFu + ((cv.u >> 16) & 1u);
  return (u16b)(r >> 16);
}

// A fragment (16x32 bf16): lane holds row m = l%16; 16 contiguous K at
// (l/16)*16.  `base` points at element (row, k0).
__device__ __forceinline__ v16bf load_a_frag(const u16b* base) {
  v8bf lo = *(const v8bf*)(base);
  v8bf hi = *(const v8bf*)(base + 8);
  return __builtin_shufflevector(lo, hi, 0, 1, 2, 3, 4, 5, 6, 7,
                                 8, 9, 10, 11, 12, 13, 14, 15);
}

// B fragment (32x16 bf16): lane holds column n = l%16; 16 contiguous K at
// (l/16)*16.  `base` points at element (k0, col) in a K-contiguous layout.
__device__ __forceinline__ v16bf load_b_frag(const u16b* base) {
  return *(const v16bf*)(base);
}

// ---------------------------------------------------------------------------
// fp32 -> bf16 conversion
// ---------------------------------------------------------------------------
__global__ void cvt_k(const float* __restrict__ in, u16b* __restrict__ out, int n) {
  int i = blockIdx.x * 256 + threadIdx.x;
  if (i < n) out[i] = f2bf(in[i]);
}

// ---------------------------------------------------------------------------
// Generic C = A @ B^T GEMM (A:[M,K] bf16 row-major, B:[N,K] bf16 row-major).
// 4 waves/block, 64x64 tile per wave -> 128x128 per block.
// MODE (compile-time):
//   0: Q: write bf16 qc=acc+bias+e0[n], qr=acc+bias+e1[n] at [b,n,i,d]
//   1: K: bf16 at [b,n,j,d]
//   2: V: bf16 TRANSPOSED at [b,n,d,j]
//   3: RK: bf16 at [n,jr,d] (rows are jr, no batch)
//   4: fp32 out = acc + bias[n] + e0[m*N+n]   (residual add)
//   5: bf16 out = relu(acc + bias[n]) at [m*N+n]
// ---------------------------------------------------------------------------
template <int MODE>
__global__ __launch_bounds__(128, 1) void gemm_bf16_k(
    const u16b* __restrict__ A, const u16b* __restrict__ B,
    int N, int K,
    const float* __restrict__ bias,
    const float* __restrict__ e0, const float* __restrict__ e1,
    void* __restrict__ o0, void* __restrict__ o1)
{
  const int lane = threadIdx.x & 31;
  const int wid  = threadIdx.x >> 5;
  const int lr = lane & 15, lh = lane >> 4;
  const int mwave = (blockIdx.y * 2 + (wid >> 1)) * 64;
  const int nwave = (blockIdx.x * 2 + (wid & 1)) * 64;

  v8f acc[4][4];
#pragma unroll
  for (int a = 0; a < 4; ++a)
#pragma unroll
    for (int b = 0; b < 4; ++b) acc[a][b] = v8f_zero();

  for (int kk = 0; kk < K; kk += 32) {
    if (kk + 32 < K) {
      __builtin_prefetch(A + (size_t)(mwave + lr) * K + kk + 32, 0, 3);
      __builtin_prefetch(B + (size_t)(nwave + lr) * K + kk + 32, 0, 3);
    }
    v16bf af[4], bf[4];
#pragma unroll
    for (int mt = 0; mt < 4; ++mt)
      af[mt] = load_a_frag(A + (size_t)(mwave + mt * 16 + lr) * K + kk + lh * 16);
#pragma unroll
    for (int nt = 0; nt < 4; ++nt)
      bf[nt] = load_b_frag(B + (size_t)(nwave + nt * 16 + lr) * K + kk + lh * 16);
#pragma unroll
    for (int mt = 0; mt < 4; ++mt)
#pragma unroll
      for (int nt = 0; nt < 4; ++nt)
        acc[mt][nt] = wmma_bf16(af[mt], bf[nt], acc[mt][nt]);
  }

#pragma unroll
  for (int mt = 0; mt < 4; ++mt) {
#pragma unroll
    for (int nt = 0; nt < 4; ++nt) {
      // hoisted per-column quantities (constant across r)
      const int n = nwave + nt * 16 + lr;
      const int h = n >> 6, d = n & 63;
      const float bn = bias[n];
      float a0 = 0.f, a1 = 0.f;
      if (MODE == 0) { a0 = bn + e0[n]; a1 = bn + e1[n]; }
#pragma unroll
      for (int r = 0; r < 8; ++r) {
        const int m = mwave + mt * 16 + r + 8 * lh;
        const float v = acc[mt][nt][r];
        if (MODE == 0) {
          const int bb = m & 3, i = m >> 2;
          const size_t idx = ((size_t)(bb * NHEAD + h) * QLEN + i) * DHEAD + d;
          ((u16b*)o0)[idx] = f2bf(v + a0);
          ((u16b*)o1)[idx] = f2bf(v + a1);
        } else if (MODE == 1) {
          const int bb = m & 3, i = m >> 2;
          ((u16b*)o0)[((size_t)(bb * NHEAD + h) * QLEN + i) * DHEAD + d] =
              f2bf(v + bn);
        } else if (MODE == 2) {
          const int bb = m & 3, i = m >> 2;
          ((u16b*)o0)[((size_t)(bb * NHEAD + h) * DHEAD + d) * QLEN + i] =
              f2bf(v + bn);
        } else if (MODE == 3) {
          ((u16b*)o0)[((size_t)h * QLEN + m) * DHEAD + d] = f2bf(v + bn);
        } else if (MODE == 4) {
          ((float*)o0)[(size_t)m * N + n] = v + bn + e0[(size_t)m * N + n];
        } else { // MODE 5
          ((u16b*)o0)[(size_t)m * N + n] = f2bf(fmaxf(v + bn, 0.0f));
        }
      }
    }
  }
}

// ---------------------------------------------------------------------------
// Fused rel-pos flash attention.  One wave per block; 32 query rows per
// block, streaming 64-key tiles.  S = 0.125*((q+rwb)K^T + BD) with
// BD[i,j] = (q+rrb)_i . rk[L-1-i+j] gathered from an LDS-staged WMMA window.
// Only qcF/Oacc/stats are persistent; qr fragments are reloaded per key tile
// (L2-resident) to keep every phase under 256 architectural VGPRs.
// ---------------------------------------------------------------------------
__global__ __launch_bounds__(32, 1) void attn_k(
    const u16b* __restrict__ qc, const u16b* __restrict__ qr,
    const u16b* __restrict__ kmat, const u16b* __restrict__ vT,
    const u16b* __restrict__ rk, u16b* __restrict__ av)
{
  __shared__ alignas(64) float ldsBD[32 * 96];
  __shared__ alignas(64) u16b  ldsP[32 * 64];

  const int lane = threadIdx.x;
  const int lr = lane & 15, lh = lane >> 4;
  const int i0 = blockIdx.x * 32;
  const int h  = blockIdx.y & (NHEAD - 1);
  const int b  = blockIdx.y >> 4;
  const size_t bn = (size_t)(b * NHEAD + h);
  const u16b* qc_p = qc + bn * QLEN * DHEAD;
  const u16b* qr_p = qr + bn * QLEN * DHEAD;
  const u16b* k_p  = kmat + bn * QLEN * DHEAD;
  const u16b* v_p  = vT + bn * DHEAD * QLEN;   // [d][j]
  const u16b* rk_p = rk + (size_t)h * QLEN * DHEAD;

  // Content-Q fragments stay resident across the whole key loop (32 VGPRs).
  v16bf qcF[2][2];
#pragma unroll
  for (int mt = 0; mt < 2; ++mt)
#pragma unroll
    for (int ks = 0; ks < 2; ++ks)
      qcF[mt][ks] = load_a_frag(qc_p + (size_t)(i0 + mt * 16 + lr) * DHEAD +
                                ks * 32 + lh * 16);

  v8f Oacc[2][4];
#pragma unroll
  for (int mt = 0; mt < 2; ++mt)
#pragma unroll
    for (int dt = 0; dt < 4; ++dt) Oacc[mt][dt] = v8f_zero();
  float mrow[2][8], lrow[2][8];
#pragma unroll
  for (int mt = 0; mt < 2; ++mt)
#pragma unroll
    for (int r = 0; r < 8; ++r) { mrow[mt][r] = -1e30f; lrow[mt][r] = 0.f; }

  const int njt = ((i0 + 31) >> 6) + 1;
  for (int jt = 0; jt < njt; ++jt) {
    const int j0 = jt * 64;
    const int jrbase = (QLEN - 1) - (i0 + 31) + j0;

    // ---- BD window (32 rows x 96 cols) in two 48-col half-passes ----
    {
      // Position-Q fragments: transient per key tile (dead after this block).
      v16bf qrF[2][2];
#pragma unroll
      for (int mt = 0; mt < 2; ++mt)
#pragma unroll
        for (int ks = 0; ks < 2; ++ks)
          qrF[mt][ks] = load_a_frag(qr_p + (size_t)(i0 + mt * 16 + lr) * DHEAD +
                                    ks * 32 + lh * 16);
#pragma unroll
      for (int half = 0; half < 2; ++half) {
        v8f bd[2][3];
#pragma unroll
        for (int mt = 0; mt < 2; ++mt)
#pragma unroll
          for (int nb = 0; nb < 3; ++nb) bd[mt][nb] = v8f_zero();
#pragma unroll
        for (int ks = 0; ks < 2; ++ks) {
          v16bf rb[3];
          const int kb0 = ks * 32 + lh * 16;
#pragma unroll
          for (int nb = 0; nb < 3; ++nb) {
            int jr = jrbase + (half * 3 + nb) * 16 + lr;
            jr = (jr < QLEN) ? jr : (QLEN - 1);  // clamp; clamped cols unused
            rb[nb] = load_b_frag(rk_p + (size_t)jr * DHEAD + kb0);
          }
#pragma unroll
          for (int mt = 0; mt < 2; ++mt)
#pragma unroll
            for (int nb = 0; nb < 3; ++nb)
              bd[mt][nb] = wmma_bf16(qrF[mt][ks], rb[nb], bd[mt][nb]);
        }
#pragma unroll
        for (int mt = 0; mt < 2; ++mt)
#pragma unroll
          for (int nb = 0; nb < 3; ++nb)
#pragma unroll
            for (int r = 0; r < 8; ++r)
              ldsBD[(mt * 16 + r + 8 * lh) * 96 + (half * 3 + nb) * 16 + lr] =
                  bd[mt][nb][r];
      }
    }
    __syncthreads();

    // ---- content scores (K fragments loaded in pairs to cap pressure) ----
    v8f sc[2][4];
#pragma unroll
    for (int mt = 0; mt < 2; ++mt)
#pragma unroll
      for (int nt = 0; nt < 4; ++nt) sc[mt][nt] = v8f_zero();
#pragma unroll
    for (int ks = 0; ks < 2; ++ks) {
      const int kb0 = ks * 32 + lh * 16;
#pragma unroll
      for (int ntp = 0; ntp < 2; ++ntp) {
        v16bf kf[2];
#pragma unroll
        for (int q = 0; q < 2; ++q)
          kf[q] = load_b_frag(
              k_p + (size_t)(j0 + (ntp * 2 + q) * 16 + lr) * DHEAD + kb0);
#pragma unroll
        for (int mt = 0; mt < 2; ++mt)
#pragma unroll
          for (int q = 0; q < 2; ++q)
            sc[mt][ntp * 2 + q] =
                wmma_bf16(qcF[mt][ks], kf[q], sc[mt][ntp * 2 + q]);
      }
    }

    // ---- add rel-shifted BD, scale, causal mask ----
#pragma unroll
    for (int mt = 0; mt < 2; ++mt)
#pragma unroll
      for (int nt = 0; nt < 4; ++nt) {
        // tile entirely below the diagonal -> no per-element mask needed
        const bool full = (j0 + nt * 16 + 15) <= (i0 + mt * 16);
        if (full) {
#pragma unroll
          for (int r = 0; r < 8; ++r) {
            const int irow = mt * 16 + r + 8 * lh;
            const float bdv = ldsBD[irow * 96 + (31 - irow + nt * 16 + lr)];
            sc[mt][nt][r] = 0.125f * (sc[mt][nt][r] + bdv);
          }
        } else {
#pragma unroll
          for (int r = 0; r < 8; ++r) {
            const int irow = mt * 16 + r + 8 * lh;
            const int j = j0 + nt * 16 + lr;
            const float bdv = ldsBD[irow * 96 + (31 - irow + nt * 16 + lr)];
            const float val = 0.125f * (sc[mt][nt][r] + bdv);
            sc[mt][nt][r] = (j <= i0 + irow) ? val : -1e30f;
          }
        }
      }

    // ---- online softmax (rows live in 16-lane halves) ----
#pragma unroll
    for (int mt = 0; mt < 2; ++mt)
#pragma unroll
      for (int r = 0; r < 8; ++r) {
        float vmax = sc[mt][0][r];
#pragma unroll
        for (int nt = 1; nt < 4; ++nt) vmax = fmaxf(vmax, sc[mt][nt][r]);
#pragma unroll
        for (int off = 8; off >= 1; off >>= 1)
          vmax = fmaxf(vmax, __shfl_xor(vmax, off, 32));
        const float mold = mrow[mt][r];
        const float mnew = fmaxf(mold, vmax);
        const float alpha = __expf(mold - mnew);
        float ps = 0.f;
#pragma unroll
        for (int nt = 0; nt < 4; ++nt) {
          const float p = __expf(sc[mt][nt][r] - mnew);
          sc[mt][nt][r] = p;
          ps += p;
        }
#pragma unroll
        for (int off = 8; off >= 1; off >>= 1) ps += __shfl_xor(ps, off, 32);
        lrow[mt][r] = lrow[mt][r] * alpha + ps;
        mrow[mt][r] = mnew;
#pragma unroll
        for (int dt = 0; dt < 4; ++dt) Oacc[mt][dt][r] *= alpha;
      }

    // ---- stage P (bf16) through LDS to reach A-fragment layout ----
#pragma unroll
    for (int mt = 0; mt < 2; ++mt)
#pragma unroll
      for (int nt = 0; nt < 4; ++nt)
#pragma unroll
        for (int r = 0; r < 8; ++r)
          ldsP[(mt * 16 + r + 8 * lh) * 64 + nt * 16 + lr] = f2bf(sc[mt][nt][r]);
    __syncthreads();

    // ---- O += P @ V ----
#pragma unroll
    for (int ks = 0; ks < 2; ++ks) {
      v16bf pf[2], vf[4];
      const int ka0 = ks * 32 + lh * 16;
      const int kb0 = ks * 32 + lh * 16;
#pragma unroll
      for (int mt = 0; mt < 2; ++mt)
        pf[mt] = load_a_frag(ldsP + (mt * 16 + lr) * 64 + ka0);
#pragma unroll
      for (int dt = 0; dt < 4; ++dt)
        vf[dt] = load_b_frag(v_p + (size_t)(dt * 16 + lr) * QLEN + j0 + kb0);
#pragma unroll
      for (int mt = 0; mt < 2; ++mt)
#pragma unroll
        for (int dt = 0; dt < 4; ++dt)
          Oacc[mt][dt] = wmma_bf16(pf[mt], vf[dt], Oacc[mt][dt]);
    }
    __syncthreads();
  }

  // ---- normalize and emit attn_vec (token-major [i*BSZ+b][n*64+d]) ----
#pragma unroll
  for (int mt = 0; mt < 2; ++mt)
#pragma unroll
    for (int r = 0; r < 8; ++r) {
      const float inv = 1.0f / lrow[mt][r];
      const int i = i0 + mt * 16 + r + 8 * lh;
      const size_t tb = ((size_t)i * BSZn + b) * DMODEL + h * DHEAD;
#pragma unroll
      for (int dt = 0; dt < 4; ++dt)
        av[tb + dt * 16 + lr] = f2bf(Oacc[mt][dt][r] * inv);
    }
}

// ---------------------------------------------------------------------------
// Row LayerNorm over DMODEL; optional bf16 copy for the next GEMM's A.
// ---------------------------------------------------------------------------
__global__ __launch_bounds__(256) void ln_k(
    const float* __restrict__ x, const float* __restrict__ g,
    const float* __restrict__ be, float* __restrict__ of,
    u16b* __restrict__ ob)
{
  __shared__ float rs[256], rq[256];
  const int row = blockIdx.x;
  const float* xr = x + (size_t)row * DMODEL;
  float s = 0.f, q = 0.f;
  for (int c = threadIdx.x; c < DMODEL; c += 256) {
    const float v = xr[c];
    s += v; q += v * v;
  }
  rs[threadIdx.x] = s; rq[threadIdx.x] = q;
  __syncthreads();
  for (int o = 128; o > 0; o >>= 1) {
    if (threadIdx.x < o) {
      rs[threadIdx.x] += rs[threadIdx.x + o];
      rq[threadIdx.x] += rq[threadIdx.x + o];
    }
    __syncthreads();
  }
  const float mu  = rs[0] * (1.0f / DMODEL);
  const float var = rq[0] * (1.0f / DMODEL) - mu * mu;
  const float rstd = rsqrtf(var + 1e-5f);
  for (int c = threadIdx.x; c < DMODEL; c += 256) {
    const float v = (xr[c] - mu) * rstd * g[c] + be[c];
    of[(size_t)row * DMODEL + c] = v;
    if (ob) ob[(size_t)row * DMODEL + c] = f2bf(v);
  }
}

// ---------------------------------------------------------------------------
extern "C" void kernel_launch(void* const* d_in, const int* in_sizes, int n_in,
                              void* d_out, int out_size, void* d_ws, size_t ws_size,
                              hipStream_t stream)
{
  (void)in_sizes; (void)n_in; (void)out_size; (void)ws_size;
  const float* w   = (const float*)d_in[0];
  const float* r   = (const float*)d_in[1];
  const float* Wq  = (const float*)d_in[3];
  const float* bq  = (const float*)d_in[4];
  const float* Wk  = (const float*)d_in[5];
  const float* bk  = (const float*)d_in[6];
  const float* Wv  = (const float*)d_in[7];
  const float* bv  = (const float*)d_in[8];
  const float* Wr  = (const float*)d_in[9];
  const float* br  = (const float*)d_in[10];
  const float* Wo  = (const float*)d_in[11];
  const float* bo  = (const float*)d_in[12];
  const float* rwb = (const float*)d_in[13];
  const float* rrb = (const float*)d_in[14];
  const float* g1  = (const float*)d_in[15];
  const float* be1 = (const float*)d_in[16];
  const float* W1  = (const float*)d_in[17];
  const float* b1  = (const float*)d_in[18];
  const float* W2  = (const float*)d_in[19];
  const float* b2  = (const float*)d_in[20];
  const float* g2  = (const float*)d_in[21];
  const float* be2 = (const float*)d_in[22];

  char* ws = (char*)d_ws;
  size_t off = 0;
  auto alloc = [&](size_t bytes) -> void* {
    off = (off + 255) & ~(size_t)255;
    void* p = ws + off;
    off += bytes;
    return p;
  };
  const size_t TOK = (size_t)QLEN * BSZn;  // 8192 tokens

  u16b* wbf = (u16b*)alloc(TOK * DMODEL * 2);
  u16b* rbf = (u16b*)alloc((size_t)QLEN * DMODEL * 2);
  u16b* wqb = (u16b*)alloc((size_t)DMODEL * DMODEL * 2);
  u16b* wkb = (u16b*)alloc((size_t)DMODEL * DMODEL * 2);
  u16b* wvb = (u16b*)alloc((size_t)DMODEL * DMODEL * 2);
  u16b* wrb = (u16b*)alloc((size_t)DMODEL * DMODEL * 2);
  u16b* wob = (u16b*)alloc((size_t)DMODEL * DMODEL * 2);
  u16b* w1b = (u16b*)alloc((size_t)DINNER * DMODEL * 2);
  u16b* w2b = (u16b*)alloc((size_t)DMODEL * DINNER * 2);
  u16b* qcb = (u16b*)alloc(TOK * DMODEL * 2);
  u16b* qrb = (u16b*)alloc(TOK * DMODEL * 2);
  u16b* kbb = (u16b*)alloc(TOK * DMODEL * 2);
  u16b* vtb = (u16b*)alloc(TOK * DMODEL * 2);
  u16b* rkb = (u16b*)alloc((size_t)NHEAD * QLEN * DHEAD * 2);
  u16b* avb = (u16b*)alloc(TOK * DMODEL * 2);
  float* x1   = (float*)alloc(TOK * DMODEL * 4);
  float* ln1f = (float*)alloc(TOK * DMODEL * 4);
  u16b* ln1b  = (u16b*)alloc(TOK * DMODEL * 2);
  u16b* h1    = (u16b*)alloc(TOK * DINNER * 2);
  float* x2 = x1;  // x1 dead after LN1 -> reuse

  auto cvt = [&](const float* src, u16b* dst, int n) {
    cvt_k<<<(n + 255) / 256, 256, 0, stream>>>(src, dst, n);
  };
  cvt(w,  wbf, (int)(TOK * DMODEL));
  cvt(r,  rbf, QLEN * DMODEL);
  cvt(Wq, wqb, DMODEL * DMODEL);
  cvt(Wk, wkb, DMODEL * DMODEL);
  cvt(Wv, wvb, DMODEL * DMODEL);
  cvt(Wr, wrb, DMODEL * DMODEL);
  cvt(Wo, wob, DMODEL * DMODEL);
  cvt(W1, w1b, DINNER * DMODEL);
  cvt(W2, w2b, DMODEL * DINNER);

  const dim3 blk(128);
  // projections
  gemm_bf16_k<0><<<dim3(DMODEL / 128, (unsigned)(TOK / 128)), blk, 0, stream>>>(
      wbf, wqb, DMODEL, DMODEL, bq, rwb, rrb, qcb, qrb);
  gemm_bf16_k<1><<<dim3(DMODEL / 128, (unsigned)(TOK / 128)), blk, 0, stream>>>(
      wbf, wkb, DMODEL, DMODEL, bk, nullptr, nullptr, kbb, nullptr);
  gemm_bf16_k<2><<<dim3(DMODEL / 128, (unsigned)(TOK / 128)), blk, 0, stream>>>(
      wbf, wvb, DMODEL, DMODEL, bv, nullptr, nullptr, vtb, nullptr);
  gemm_bf16_k<3><<<dim3(DMODEL / 128, QLEN / 128), blk, 0, stream>>>(
      rbf, wrb, DMODEL, DMODEL, br, nullptr, nullptr, rkb, nullptr);

  // fused rel-pos flash attention
  attn_k<<<dim3(QLEN / 32, BSZn * NHEAD), dim3(32), 0, stream>>>(
      qcb, qrb, kbb, vtb, rkb, avb);

  // output projection + residual, LN1
  gemm_bf16_k<4><<<dim3(DMODEL / 128, (unsigned)(TOK / 128)), blk, 0, stream>>>(
      avb, wob, DMODEL, DMODEL, bo, w, nullptr, x1, nullptr);
  ln_k<<<(unsigned)TOK, 256, 0, stream>>>(x1, g1, be1, ln1f, ln1b);

  // FFN
  gemm_bf16_k<5><<<dim3(DINNER / 128, (unsigned)(TOK / 128)), blk, 0, stream>>>(
      ln1b, w1b, DINNER, DMODEL, b1, nullptr, nullptr, h1, nullptr);
  gemm_bf16_k<4><<<dim3(DMODEL / 128, (unsigned)(TOK / 128)), blk, 0, stream>>>(
      h1, w2b, DMODEL, DINNER, b2, ln1f, nullptr, x2, nullptr);
  ln_k<<<(unsigned)TOK, 256, 0, stream>>>(x2, g2, be2, (float*)d_out, nullptr);
}
